// NetGIN_52149492908641
// MI455X (gfx1250) — matrix operand order repaired and checked by reading, not compiled, not run on hardware
//
#include <hip/hip_runtime.h>
#include <math.h>

#define N_NODES 100000
#define N_EDGES 800000
#define NFEAT   83
#define DDIM    128
#define BGRAPH  256
#define TILE_K  32

typedef __attribute__((ext_vector_type(16))) _Float16 v16h;
typedef __attribute__((ext_vector_type(8)))  float    v8f;
typedef __attribute__((ext_vector_type(4)))  _Float16 h4v;
typedef __attribute__((ext_vector_type(8)))  _Float16 h8v;

__device__ __forceinline__ v8f wmma16(v16h a, v16h b, v8f c) {
    return __builtin_amdgcn_wmma_f32_16x16x32_f16(false, a, false, b, (short)0, c,
                                                  false, false);
}
__device__ __forceinline__ v16h cat8(h8v lo, h8v hi) {
    return __builtin_shufflevector(lo, hi, 0,1,2,3,4,5,6,7,8,9,10,11,12,13,14,15);
}

// ---------------- generic fill ----------------
__global__ void k_fill(float* __restrict__ p, float v, long long n) {
    long long i = (long long)blockIdx.x * blockDim.x + threadIdx.x;
    if (i < n) p[i] = v;
}

// ---------------- weight fp32 -> f16, column-major, K padded to Kpad ----------------
// Wh[n * Kpad + k] = (f16) W[k * 128 + n]   (zero padded for k >= K)
__global__ void k_w2h(const float* __restrict__ W, _Float16* __restrict__ Wh,
                      int K, int Kpad) {
    int idx = blockIdx.x * blockDim.x + threadIdx.x;
    if (idx >= 128 * Kpad) return;
    int n = idx / Kpad, k = idx % Kpad;
    Wh[idx] = (k < K) ? (_Float16)W[(long long)k * 128 + n] : (_Float16)0.f;
}

// ---------------- WMMA GEMM: C[M x 128] = A[M x K] @ W[K x 128] + bias ----------------
// K compile-time (83/128/256/512); lda == K. 256 threads = 8 waves; block covers
// 32 rows x 128 cols; each wave computes two 16x16 tiles sharing one B fragment.
// A fp32 row-major staged to LDS as f16; B pre-converted f16 column-major in global
// (<=128KB, L2 resident, loaded straight into fragments with global_load_b128).
template<int K>
__global__ void k_gemm(const float* __restrict__ A, const _Float16* __restrict__ Wh,
                       const float* __restrict__ bias, float* __restrict__ C,
                       int ldc, int act) {
    constexpr int KP = (K + TILE_K - 1) / TILE_K * TILE_K;
    __shared__ _Float16 sA[32 * TILE_K];    // [row][k]  32x32

    const int t    = threadIdx.x;
    const int lane = t & 31;
    const int wave = t >> 5;       // 0..7  -> column tile
    const int half = lane >> 4;    // lane group 0/1
    const int l16  = lane & 15;
    const int rowBase = blockIdx.x * 32;

    const int srow = t >> 3;                 // staging: row 0..31
    const int sk   = (t & 7) * 4;            // staging: k quad 0,4,..,28
    const float* Arow = A + (long long)(rowBase + srow) * K + sk;
    // B fragment: lane l16 -> column, lane group h holds K{8h..8h+7, 16+8h..16+8h+7}
    const _Float16* Bcol = Wh + (long long)(wave * 16 + l16) * KP + half * 8;

    v8f acc0 = {}, acc1 = {};

#pragma unroll
    for (int kk = 0; kk < K; kk += TILE_K) {
        // ---- stage A tile (32x32) as f16 ----
        h4v hp;
        if constexpr (K % 4 == 0) {
            float4 f = *(const float4*)&Arow[kk];          // global_load_b128
            hp[0] = (_Float16)f.x; hp[1] = (_Float16)f.y;
            hp[2] = (_Float16)f.z; hp[3] = (_Float16)f.w;
        } else {
            if (kk + TILE_K <= K) {
#pragma unroll
                for (int j = 0; j < 4; ++j) hp[j] = (_Float16)Arow[kk + j];
            } else {                                        // static tail (K=83)
#pragma unroll
                for (int j = 0; j < 4; ++j)
                    hp[j] = (kk + sk + j < K) ? (_Float16)Arow[kk + j] : (_Float16)0.f;
            }
        }
        if (kk + TILE_K < K)
            __builtin_prefetch(&Arow[kk + TILE_K], 0, 1);   // global_prefetch_b8
        *(h4v*)&sA[srow * TILE_K + sk] = hp;                // ds_store_b64
        __syncthreads();

        // A 16x32 f16 layout: lanes 0-15 K{0..7,16..23}, lanes 16-31 K{8..15,24..31}
        h8v a00 = *(const h8v*)&sA[l16 * TILE_K + half * 8];
        h8v a01 = *(const h8v*)&sA[l16 * TILE_K + 16 + half * 8];
        h8v a10 = *(const h8v*)&sA[(16 + l16) * TILE_K + half * 8];
        h8v a11 = *(const h8v*)&sA[(16 + l16) * TILE_K + 16 + half * 8];
        // B 32x16 f16 layout: lanes 0-15 K{0..7,16..23}, lanes 16-31 K{8..15,24..31}
        h8v b0 = *(const h8v*)&Bcol[kk];
        h8v b1 = *(const h8v*)&Bcol[kk + 16];
        v16h bf  = cat8(b0, b1);
        v16h af0 = cat8(a00, a01);
        v16h af1 = cat8(a10, a11);
        acc0 = wmma16(af0, bf, acc0);       // two independent chains share bf
        acc1 = wmma16(af1, bf, acc1);
        __syncthreads();
    }

    const int col = wave * 16 + l16;
    const float bv = bias[col];
    // C/D f32 16x16 layout: lanes 0-15 -> M=r ; lanes 16-31 -> M=r+8
#pragma unroll
    for (int r = 0; r < 8; ++r) {
        int r0 = rowBase + r + half * 8;
        float v0 = acc0[r] + bv;
        float v1 = acc1[r] + bv;
        if (act) { v0 = fmaxf(v0, 0.f); v1 = fmaxf(v1, 0.f); }
        C[(long long)r0 * ldc + col] = v0;
        C[(long long)(r0 + 16) * ldc + col] = v1;
    }
}

// ---------------- column sums / sums-of-squares for BN (C == 128) ----------------
__global__ void k_colstats(const float* __restrict__ x, float* __restrict__ stats, int M) {
    int c = threadIdx.x;                         // blockDim == 128
    int chunk = (M + gridDim.x - 1) / gridDim.x;
    int r0 = blockIdx.x * chunk;
    int r1 = r0 + chunk; if (r1 > M) r1 = M;
    float s = 0.f, s2 = 0.f;
    for (int r = r0; r < r1; ++r) {
        float v = x[(long long)r * 128 + c];
        s += v; s2 += v * v;
    }
    atomicAdd(&stats[c], s);
    atomicAdd(&stats[128 + c], s2);
}

// ---------------- BN (batch stats, biased var) + ReLU ----------------
__global__ void k_bn_relu(const float* __restrict__ x, const float* __restrict__ stats,
                          const float* __restrict__ g, const float* __restrict__ be,
                          float* __restrict__ out, int M, int ostride) {
    long long i = (long long)blockIdx.x * blockDim.x + threadIdx.x;
    if (i >= (long long)M * 128) return;
    int r = (int)(i >> 7), c = (int)(i & 127);
    float mu  = stats[c] / (float)M;
    float var = stats[128 + c] / (float)M - mu * mu;
    float v = (x[i] - mu) * rsqrtf(var + 1e-5f) * g[c] + be[c];
    out[(long long)r * ostride + c] = fmaxf(v, 0.f);
}

// ---------------- edge scatter-add: agg[dst] += x[src] ----------------
__global__ void k_scatter(const float* __restrict__ x, const int* __restrict__ ei,
                          float* __restrict__ agg, int E, int C) {
    int e = blockIdx.x;
    if (e >= E) return;
    int src = ei[e];
    int dst = ei[E + e];
    const float* xs = x + (long long)src * C;
    float* ad = agg + (long long)dst * C;
    for (int c = threadIdx.x; c < C; c += blockDim.x)
        atomicAdd(&ad[c], xs[c]);
}

// ---------------- GIN combine (in-place): agg = (1+eps)*x + agg ----------------
__global__ void k_gin_combine(float* __restrict__ agg, const float* __restrict__ x,
                              const float* __restrict__ eps, long long n) {
    long long i = (long long)blockIdx.x * blockDim.x + threadIdx.x;
    if (i < n) agg[i] = (1.f + eps[0]) * x[i] + agg[i];
}

// ---------------- LSTM gates: gates = qstar@Wih^T + bih + h@Whh^T + bhh ----------------
__global__ void k_lstm_gates(const float* __restrict__ qs, const float* __restrict__ h,
                             const float* __restrict__ Wih, const float* __restrict__ Whh,
                             const float* __restrict__ bih, const float* __restrict__ bhh,
                             float* __restrict__ gates) {
    int idx = blockIdx.x * blockDim.x + threadIdx.x;   // BGRAPH*512
    if (idx >= BGRAPH * 512) return;
    int b = idx >> 9, j = idx & 511;
    const float* q  = qs + (long long)b * 256;
    const float* hr = h + (long long)b * 128;
    const float* wi = Wih + (long long)j * 256;
    const float* wh = Whh + (long long)j * 128;
    float s = bih[j] + bhh[j];
    for (int k = 0; k < 256; ++k) s += q[k] * wi[k];
    for (int k = 0; k < 128; ++k) s += hr[k] * wh[k];
    gates[idx] = s;
}

__device__ __forceinline__ float sigm(float x) { return 1.f / (1.f + expf(-x)); }

// ---------------- LSTM state update; also writes q-part of q_star ----------------
__global__ void k_lstm_update(const float* __restrict__ gates, float* __restrict__ c,
                              float* __restrict__ h, float* __restrict__ qs) {
    int idx = blockIdx.x * blockDim.x + threadIdx.x;   // BGRAPH*128
    if (idx >= BGRAPH * 128) return;
    int b = idx >> 7, k = idx & 127;
    const float* gr = gates + (long long)b * 512;
    float ig = sigm(gr[k]);
    float fg = sigm(gr[128 + k]);
    float gg = tanhf(gr[256 + k]);
    float og = sigm(gr[384 + k]);
    float cn = fg * c[idx] + ig * gg;
    c[idx] = cn;
    float hn = og * tanhf(cn);
    h[idx] = hn;
    qs[(long long)b * 256 + k] = hn;
}

// ---------------- zero the r-part of q_star ----------------
__global__ void k_zero_r(float* __restrict__ qs) {
    int idx = blockIdx.x * blockDim.x + threadIdx.x;   // BGRAPH*128
    if (idx >= BGRAPH * 128) return;
    qs[(long long)(idx >> 7) * 256 + 128 + (idx & 127)] = 0.f;
}

// ---------------- attention logits: e[n] = dot(x[n], q[batch[n]]) ----------------
__global__ void k_attn_e(const float* __restrict__ x, const float* __restrict__ q,
                         const int* __restrict__ batch, float* __restrict__ e, int N) {
    int n = blockIdx.x * blockDim.x + threadIdx.x;
    if (n >= N) return;
    const float* xr = x + (long long)n * 128;
    const float* qr = q + (long long)batch[n] * 128;
    float s = 0.f;
    for (int c = 0; c < 128; ++c) s += xr[c] * qr[c];
    e[n] = s;
}

__device__ __forceinline__ void atomicMaxF(float* addr, float val) {
    unsigned int* a = (unsigned int*)addr;
    unsigned int old = *a;
    while (val > __uint_as_float(old)) {
        unsigned int assumed = old;
        old = atomicCAS(a, assumed, __float_as_uint(val));
        if (old == assumed) break;
    }
}

__global__ void k_segmax(const float* __restrict__ e, const int* __restrict__ batch,
                         float* __restrict__ emax, int N) {
    int n = blockIdx.x * blockDim.x + threadIdx.x;
    if (n < N) atomicMaxF(&emax[batch[n]], e[n]);
}

__global__ void k_fixmax(float* __restrict__ emax, int B) {
    int b = threadIdx.x;
    if (b < B) { float v = emax[b]; if (v < -3.0e38f) emax[b] = 0.f; }
}

__global__ void k_attn_a(const float* __restrict__ e, const float* __restrict__ emax,
                         const int* __restrict__ batch, float* __restrict__ a,
                         float* __restrict__ asum, int N) {
    int n = blockIdx.x * blockDim.x + threadIdx.x;
    if (n >= N) return;
    int b = batch[n];
    float av = expf(e[n] - emax[b]);
    a[n] = av;
    atomicAdd(&asum[b], av);
}

// ---------------- r[b] += (a[n]/asum[b]) * x[n] into r-part of q_star ----------------
__global__ void k_attn_r(const float* __restrict__ x, const float* __restrict__ a,
                         const float* __restrict__ asum, const int* __restrict__ batch,
                         float* __restrict__ qs, int N) {
    long long i = (long long)blockIdx.x * blockDim.x + threadIdx.x;
    if (i >= (long long)N * 128) return;
    int n = (int)(i >> 7), c = (int)(i & 127);
    int b = batch[n];
    float w = a[n] / (asum[b] + 1e-16f);
    atomicAdd(&qs[(long long)b * 256 + 128 + c], w * x[i]);
}

// ---------------- concat [pooled_con | pooled_unc] ----------------
__global__ void k_concat(const float* __restrict__ pc, const float* __restrict__ pu,
                         float* __restrict__ out) {
    int idx = blockIdx.x * blockDim.x + threadIdx.x;   // BGRAPH*512
    if (idx >= BGRAPH * 512) return;
    int b = idx >> 9, c = idx & 511;
    out[idx] = (c < 256) ? pc[(long long)b * 256 + c] : pu[(long long)b * 256 + (c - 256)];
}

// ---------------- fc4: out[b,j] = f2[b]·W[:,j] + bias[j], 12 outputs ----------------
__global__ void k_fc4(const float* __restrict__ x, const float* __restrict__ W,
                      const float* __restrict__ bias, float* __restrict__ out, int B) {
    int idx = blockIdx.x * blockDim.x + threadIdx.x;
    if (idx >= B * 12) return;
    int b = idx / 12, j = idx % 12;
    const float* xr = x + (long long)b * 128;
    float s = bias[j];
    for (int k = 0; k < 128; ++k) s += xr[k] * W[(long long)k * 12 + j];
    out[idx] = s;
}

extern "C" void kernel_launch(void* const* d_in, const int* in_sizes, int n_in,
                              void* d_out, int out_size, void* d_ws, size_t ws_size,
                              hipStream_t stream) {
    (void)in_sizes; (void)n_in; (void)out_size; (void)ws_size;
    const int N = N_NODES, E = N_EDGES, B = BGRAPH;

    auto F = [&](int i) { return (const float*)d_in[i]; };
    auto I = [&](int i) { return (const int*)d_in[i]; };

    const float* x_unc = F(0);
    const float* x_con = F(1);
    // params flattened in dict order:
    //  gin[l][r]: {eps, W1,b1,g1,be1, W2,b2,g2,be2}  -> 9 each, 12 convs
    const int GIN0 = 2;
    const int MIX0 = GIN0 + 6 * 2 * 9;   // 110: mix[l] {W1,b1,g1,be1,W2,b2,g2,be2}
    const int S2SU = MIX0 + 6 * 8;       // 158: {Wih,Whh,bih,bhh}
    const int S2SC = S2SU + 4;           // 162
    const int FUSE = S2SC + 4;           // 166 (mlp, 8)
    const int FC1  = FUSE + 8;           // 174 {W,b}
    const int FC4  = FC1 + 2;            // 176 {W,b}
    const int* ei0       = I(FC4 + 2);   // 178: edge_index_0_con (2,E)
    const int* ei1       = I(179);
    const int* batch_unc = I(180);
    const int* batch_con = I(181);

    // ---------- workspace layout ----------
    float* ws = (float*)d_ws;
    size_t cur = 0;
    auto alloc = [&](size_t nf) { float* p = ws + cur; cur += nf; return p; };
    const size_t NB = (size_t)N * DDIM;
    float* bufX   = alloc(NB);            // current node activations
    float* bufAgg = alloc(NB);            // aggregation / GIN combine
    float* bufT   = alloc(NB);            // MLP pre-BN temp
    float* bufH   = alloc(NB);            // MLP hidden
    float* bufCat = alloc(2 * NB);        // [a | b] concat, N x 256
    float* stats  = alloc(256);
    _Float16* Wh  = (_Float16*)alloc(128 * 512 / 2);   // f16 weights, col-major, Kpad<=512
    float* qsU    = alloc((size_t)B * 256);
    float* qsC    = alloc((size_t)B * 256);
    float* hb     = alloc((size_t)B * 128);
    float* cb     = alloc((size_t)B * 128);
    float* gates  = alloc((size_t)B * 512);
    float* ebuf   = alloc((size_t)N);
    float* abuf   = alloc((size_t)N);
    float* emax   = alloc((size_t)B);
    float* asum   = alloc((size_t)B);
    float* fin    = alloc((size_t)B * 512);
    float* f1     = alloc((size_t)B * 128);
    float* f2     = alloc((size_t)B * 128);

    auto fill = [&](float* p, float v, long long n) {
        k_fill<<<dim3((unsigned)((n + 255) / 256)), 256, 0, stream>>>(p, v, n);
    };
    // lda == K in every call; M % 32 == 0 in every call (100000, 256)
    auto gemm = [&](const float* A, int M, int K, const float* W,
                    const float* b, float* C, int ldc, int act) {
        int Kpad = (K + 31) & ~31;
        k_w2h<<<dim3((128 * Kpad + 255) / 256), 256, 0, stream>>>(W, Wh, K, Kpad);
        dim3 g((unsigned)(M / 32));
        switch (K) {
            case 83:  k_gemm<83> <<<g, 256, 0, stream>>>(A, Wh, b, C, ldc, act); break;
            case 128: k_gemm<128><<<g, 256, 0, stream>>>(A, Wh, b, C, ldc, act); break;
            case 256: k_gemm<256><<<g, 256, 0, stream>>>(A, Wh, b, C, ldc, act); break;
            case 512: k_gemm<512><<<g, 256, 0, stream>>>(A, Wh, b, C, ldc, act); break;
        }
    };
    // Linear->BN->ReLU->Linear->BN->ReLU ; pb = index of W1
    auto mlp = [&](const float* in, int M, int K, int pb, float* out, int ostride) {
        long long n = (long long)M * 128;
        unsigned gn = (unsigned)((n + 255) / 256);
        gemm(in, M, K, F(pb + 0), F(pb + 1), bufT, 128, 0);
        fill(stats, 0.f, 256);
        k_colstats<<<dim3(256), 128, 0, stream>>>(bufT, stats, M);
        k_bn_relu<<<dim3(gn), 256, 0, stream>>>(bufT, stats, F(pb + 2), F(pb + 3), bufH, M, 128);
        gemm(bufH, M, 128, F(pb + 4), F(pb + 5), bufT, 128, 0);
        fill(stats, 0.f, 256);
        k_colstats<<<dim3(256), 128, 0, stream>>>(bufT, stats, M);
        k_bn_relu<<<dim3(gn), 256, 0, stream>>>(bufT, stats, F(pb + 6), F(pb + 7), out, M, ostride);
    };
    // one GIN conv: base = index of eps; out = column slice of bufCat (stride 256)
    auto gin = [&](const float* x, int C, const int* ei, int base, float* out) {
        long long n = (long long)N * C;
        fill(bufAgg, 0.f, n);
        k_scatter<<<dim3(E), 128, 0, stream>>>(x, ei, bufAgg, E, C);
        k_gin_combine<<<dim3((unsigned)((n + 255) / 256)), 256, 0, stream>>>(bufAgg, x, F(base), n);
        mlp(bufAgg, N, C, base + 1, out, 256);
    };
    auto s2s = [&](const float* x, const int* batch, int lb, float* qs) {
        fill(qs, 0.f, (long long)B * 256);
        fill(hb, 0.f, (long long)B * 128);
        fill(cb, 0.f, (long long)B * 128);
        for (int s = 0; s < 6; ++s) {
            k_lstm_gates<<<dim3(B * 512 / 256), 256, 0, stream>>>(qs, hb, F(lb + 0), F(lb + 1),
                                                                  F(lb + 2), F(lb + 3), gates);
            k_lstm_update<<<dim3(B * 128 / 128), 128, 0, stream>>>(gates, cb, hb, qs);
            fill(emax, -INFINITY, B);
            fill(asum, 0.f, B);
            k_zero_r<<<dim3(B * 128 / 256), 256, 0, stream>>>(qs);
            k_attn_e<<<dim3((N + 255) / 256), 256, 0, stream>>>(x, hb, batch, ebuf, N);
            k_segmax<<<dim3((N + 255) / 256), 256, 0, stream>>>(ebuf, batch, emax, N);
            k_fixmax<<<dim3(1), B, 0, stream>>>(emax, B);
            k_attn_a<<<dim3((N + 255) / 256), 256, 0, stream>>>(ebuf, emax, batch, abuf, asum, N);
            long long n = (long long)N * 128;
            k_attn_r<<<dim3((unsigned)((n + 255) / 256)), 256, 0, stream>>>(x, abuf, asum, batch, qs, N);
        }
    };

    // ---------- unc branch: MLP(gin[0][0].mlp) then Set2Set ----------
    mlp(x_unc, N, NFEAT, GIN0 + 1, bufX, 128);
    s2s(bufX, batch_unc, S2SU, qsU);

    // ---------- con branch: 6 dual-relation GIN blocks ----------
    const float* xcur = x_con;
    int C = NFEAT;
    for (int l = 0; l < 6; ++l) {
        gin(xcur, C, ei0, GIN0 + (l * 2 + 0) * 9, bufCat);
        gin(xcur, C, ei1, GIN0 + (l * 2 + 1) * 9, bufCat + 128);
        mlp(bufCat, N, 256, MIX0 + l * 8, bufX, 128);
        xcur = bufX;
        C = DDIM;
    }
    s2s(bufX, batch_con, S2SC, qsC);

    // ---------- fuse + heads ----------
    k_concat<<<dim3(B * 512 / 256), 256, 0, stream>>>(qsC, qsU, fin);
    mlp(fin, B, 512, FUSE, f1, 128);
    gemm(f1, B, 128, F(FC1), F(FC1 + 1), f2, 128, 1);                // fc1 + ReLU
    k_fc4<<<dim3((B * 12 + 255) / 256), 256, 0, stream>>>(f2, F(FC4), F(FC4 + 1),
                                                          (float*)d_out, B);
}
